// HybridQuantumClassifier_22565758173819
// MI455X (gfx1250) — compile-verified, matrix operand order checked by reading, not run on hardware
//
#include <hip/hip_runtime.h>

typedef __attribute__((ext_vector_type(16))) _Float16 v16h;
typedef __attribute__((ext_vector_type(8)))  float    v8f;

#define NQ 8
#define NL 4

struct c32 { float x, y; };
__device__ __forceinline__ c32 cmul(c32 a, c32 b) { return {a.x*b.x - a.y*b.y, a.x*b.y + a.y*b.x}; }
__device__ __forceinline__ c32 cadd(c32 a, c32 b) { return {a.x + b.x, a.y + b.y}; }
__device__ __forceinline__ c32 csel(int c, c32 a, c32 b) { c32 r; r.x = c ? a.x : b.x; r.y = c ? a.y : b.y; return r; }

__device__ __forceinline__ float shflx(float v, int mask) {
  return __shfl_xor(v, mask, 32);
}

__device__ __forceinline__ v8f wmma32(v16h a, v16h b, v8f c) {
  // D = A(16x32 f16) * B(32x16 f16) + C(16x16 f32)
  return __builtin_amdgcn_wmma_f32_16x16x32_f16(false, a, false, b, (short)0, c, false, false);
}

// Gather a 16x32 f16 fragment (ISA 7.12.2 16-bit A layout) from row-major LDS.
// lane L: M = L%16; VGPR v: K = k0 + (v/4)*16 + (L/16)*8 + (v%4)*2 (+1)
__device__ __forceinline__ v16h load_frag(const _Float16* p, int stride, int k0, int lane) {
  int m  = lane & 15;
  int kg = (lane >> 4) << 3;
  v16h f;
#pragma unroll
  for (int v = 0; v < 8; ++v) {
    int k = k0 + ((v >> 2) << 4) + kg + ((v & 3) << 1);
    f[2 * v]     = p[m * stride + k];
    f[2 * v + 1] = p[m * stride + k + 1];
  }
  return f;
}

// Single-qubit rotation on amplitude-bit p. p<3: register bit; p>=3: lane bit.
__device__ __forceinline__ void apply1q(c32* amp, int p, const c32* m, int lane) {
  c32 m00 = m[0], m01 = m[1], m10 = m[2], m11 = m[3];
  if (p < 3) {
    int st = 1 << p;
    c32 na[8];
#pragma unroll
    for (int j = 0; j < 8; ++j) {
      c32 lo = amp[j & ~st], hi = amp[j | st];
      na[j] = ((j >> p) & 1) ? cadd(cmul(m10, lo), cmul(m11, hi))
                             : cadd(cmul(m00, lo), cmul(m01, hi));
    }
#pragma unroll
    for (int j = 0; j < 8; ++j) amp[j] = na[j];
  } else {
    int mk  = 1 << (p - 3);
    int bit = (lane >> (p - 3)) & 1;
#pragma unroll
    for (int j = 0; j < 8; ++j) {
      c32 mine = amp[j];
      c32 oth; oth.x = shflx(mine.x, mk); oth.y = shflx(mine.y, mk);
      c32 hi = csel(bit, mine, oth);
      c32 lo = csel(bit, oth, mine);
      c32 r0 = cadd(cmul(m00, lo), cmul(m01, hi));
      c32 r1 = cadd(cmul(m10, lo), cmul(m11, hi));
      amp[j] = csel(bit, r1, r0);
    }
  }
}

// CNOT: if control-bit (pc) set, flip target-bit (pt). Pure permutation.
__device__ __forceinline__ void applycx(c32* amp, int pc, int pt, int lane) {
  if (pt < 3) {
    int st = 1 << pt;
    c32 na[8];
#pragma unroll
    for (int j = 0; j < 8; ++j) {
      int cbit = ((((lane << 3) | j) >> pc) & 1);
      na[j] = csel(cbit, amp[j ^ st], amp[j]);
    }
#pragma unroll
    for (int j = 0; j < 8; ++j) amp[j] = na[j];
  } else {
    int mk = 1 << (pt - 3);
#pragma unroll
    for (int j = 0; j < 8; ++j) {
      c32 oth; oth.x = shflx(amp[j].x, mk); oth.y = shflx(amp[j].y, mk);
      int cbit = ((((lane << 3) | j) >> pc) & 1);
      amp[j] = csel(cbit, oth, amp[j]);
    }
  }
}

__global__ __launch_bounds__(128)
void hqc_fused_kernel(const float* __restrict__ x,
                      const float* __restrict__ w1, const float* __restrict__ b1,
                      const float* __restrict__ w2, const float* __restrict__ b2,
                      const float* __restrict__ w3, const float* __restrict__ b3,
                      const float* __restrict__ qw,
                      const float* __restrict__ p1, const float* __restrict__ bp1,
                      const float* __restrict__ p2, const float* __restrict__ bp2,
                      const float* __restrict__ p3, const float* __restrict__ bp3,
                      float* __restrict__ out)
{
  // Transposed f16 weights: [N][Kpad], zero padded.
  __shared__ _Float16 s_w1t[64 * 64];
  __shared__ _Float16 s_w2t[32 * 64];
  __shared__ _Float16 s_w3t[16 * 32];
  __shared__ _Float16 s_p1t[32 * 32];
  __shared__ _Float16 s_p2t[16 * 32];
  __shared__ _Float16 s_p3t[16 * 32];
  __shared__ float s_b1[64], s_b2[32], s_b3[16], s_bp1[32], s_bp2[16], s_bp3[16];
  __shared__ c32  s_rot[NL * NQ * 4];                 // 32 gates x {m00,m01,m10,m11}
  __shared__ _Float16 s_act[4 * 16 * 64];             // per-wave activation tile (stride 64)
  __shared__ float    s_qin[4 * 16 * 8];              // per-wave tanh outputs

  const int tid = threadIdx.x;

  auto loadT = [&](_Float16* dst, const float* src, int K, int N, int Kp, int Np) {
    for (int i = tid; i < Np * Kp; i += (int)blockDim.x) {
      int n = i / Kp, k = i - n * Kp;
      dst[i] = (k < K && n < N) ? (_Float16)src[k * N + n] : (_Float16)0.f;
    }
  };
  auto loadB = [&](float* dst, const float* src, int N, int Np) {
    for (int i = tid; i < Np; i += (int)blockDim.x) dst[i] = (i < N) ? src[i] : 0.f;
  };

  loadT(s_w1t, w1, 41, 64, 64, 64);
  loadT(s_w2t, w2, 64, 32, 64, 32);
  loadT(s_w3t, w3, 32,  8, 32, 16);
  loadT(s_p1t, p1,  8, 32, 32, 32);
  loadT(s_p2t, p2, 32, 16, 32, 16);
  loadT(s_p3t, p3, 16,  5, 32, 16);
  loadB(s_b1,  b1, 64, 64);
  loadB(s_b2,  b2, 32, 32);
  loadB(s_b3,  b3,  8, 16);
  loadB(s_bp1, bp1, 32, 32);
  loadB(s_bp2, bp2, 16, 16);
  loadB(s_bp3, bp3,  5, 16);

  if (tid < NL * NQ) {
    // rot = RZ(omega) RY(theta) RZ(phi) style single-qubit matrix (ref formula)
    float phi = qw[tid * 3 + 0], th = qw[tid * 3 + 1], om = qw[tid * 3 + 2];
    float c = cosf(0.5f * th), s = sinf(0.5f * th);
    float A = 0.5f * (phi + om), Bv = 0.5f * (phi - om);
    float cA = cosf(A), sA = sinf(A), cB = cosf(Bv), sB = sinf(Bv);
    c32* r = &s_rot[tid * 4];
    r[0] = { cA * c, -sA * c};   // m00 = e^{-i(phi+om)/2} cos
    r[1] = {-cB * s, -sB * s};   // m01 = -e^{ i(phi-om)/2} sin
    r[2] = { cB * s, -sB * s};   // m10 =  e^{-i(phi-om)/2} sin
    r[3] = { cA * c,  sA * c};   // m11 =  e^{ i(phi+om)/2} cos
  }
  __syncthreads();

  const int wave = tid >> 5;
  const int lane = tid & 31;
  const int row0 = blockIdx.x * 64 + wave * 16;
  _Float16* act = s_act + wave * 16 * 64;
  float*    qin = s_qin + wave * 16 * 8;

  const int nl = lane & 15;            // C-frag N
  const int mb = (lane >> 4) << 3;     // C-frag M base (VGPR v -> M = mb+v)

  // ---- stage x tile (16 x 41 -> f16, zero padded to K=64) ----
  for (int i = lane; i < 16 * 64; i += 32) act[i] = (_Float16)0.f;
  for (int i = lane; i < 16 * 41; i += 32) {
    int r = i / 41, cc = i - r * 41;
    act[r * 64 + cc] = (_Float16)x[(row0 + r) * 41 + cc];
  }

  // ---- pre layer 1: (16x64k) x (64k x 64) -> relu ----
  {
    v8f t[4];
#pragma unroll
    for (int nt = 0; nt < 4; ++nt) {
      v8f acc = {};
#pragma unroll
      for (int k0 = 0; k0 < 64; k0 += 32)
        acc = wmma32(load_frag(act, 64, k0, lane),
                     load_frag(s_w1t + nt * 16 * 64, 64, k0, lane), acc);
      t[nt] = acc;
    }
#pragma unroll
    for (int nt = 0; nt < 4; ++nt)
#pragma unroll
      for (int v = 0; v < 8; ++v) {
        float val = t[nt][v] + s_b1[nt * 16 + nl];
        act[(mb + v) * 64 + nt * 16 + nl] = (_Float16)fmaxf(val, 0.f);
      }
  }

  // ---- pre layer 2: K=64, N=32 -> relu ----
  {
    v8f t[2];
#pragma unroll
    for (int nt = 0; nt < 2; ++nt) {
      v8f acc = {};
#pragma unroll
      for (int k0 = 0; k0 < 64; k0 += 32)
        acc = wmma32(load_frag(act, 64, k0, lane),
                     load_frag(s_w2t + nt * 16 * 64, 64, k0, lane), acc);
      t[nt] = acc;
    }
#pragma unroll
    for (int nt = 0; nt < 2; ++nt)
#pragma unroll
      for (int v = 0; v < 8; ++v) {
        float val = t[nt][v] + s_b2[nt * 16 + nl];
        act[(mb + v) * 64 + nt * 16 + nl] = (_Float16)fmaxf(val, 0.f);
      }
  }

  // ---- pre layer 3: K=32, N=8(pad16) -> tanh -> q_in ----
  {
    v8f acc = {};
    acc = wmma32(load_frag(act, 64, 0, lane), load_frag(s_w3t, 32, 0, lane), acc);
#pragma unroll
    for (int v = 0; v < 8; ++v) {
      float val = acc[v] + s_b3[nl];
      if (nl < 8) qin[(mb + v) * 8 + nl] = tanhf(val);
    }
  }

  // ---- quantum layer: one row per iteration, state in registers ----
#pragma unroll 1
  for (int row = 0; row < 16; ++row) {
    float cv[8], sv[8];
#pragma unroll
    for (int i = 0; i < 8; ++i) {
      float h = qin[row * 8 + i] * 1.5707963267948966f;
      cv[i] = cosf(h); sv[i] = sinf(h);
    }
    c32 amp[8];
#pragma unroll
    for (int j = 0; j < 8; ++j) {
      int a = (lane << 3) | j;
      float p = 1.f;
#pragma unroll
      for (int i = 0; i < 8; ++i) p *= ((a >> (7 - i)) & 1) ? sv[i] : cv[i];
      amp[j].x = p; amp[j].y = 0.f;
    }
#pragma unroll
    for (int l = 0; l < NL; ++l) {
#pragma unroll
      for (int w = 0; w < NQ; ++w)
        apply1q(amp, 7 - w, &s_rot[(l * NQ + w) * 4], lane);
      const int r = l + 1;  // l % (NQ-1) + 1
#pragma unroll
      for (int w = 0; w < NQ; ++w)
        applycx(amp, 7 - w, 7 - ((w + r) & 7), lane);
    }
    // <Z_w> = sum_a |amp|^2 * (1 - 2*bit_{7-w}(a))
    float zp[8];
#pragma unroll
    for (int w = 0; w < 8; ++w) zp[w] = 0.f;
#pragma unroll
    for (int j = 0; j < 8; ++j) {
      int a = (lane << 3) | j;
      float pr = amp[j].x * amp[j].x + amp[j].y * amp[j].y;
#pragma unroll
      for (int w = 0; w < 8; ++w) zp[w] += ((a >> (7 - w)) & 1) ? -pr : pr;
    }
#pragma unroll
    for (int mask = 1; mask < 32; mask <<= 1)
#pragma unroll
      for (int w = 0; w < 8; ++w) zp[w] += shflx(zp[w], mask);
    float zv = 0.f;
#pragma unroll
    for (int w = 0; w < 8; ++w) zv = (lane == w) ? zp[w] : zv;
    // cols 0..7 = z, cols 8..31 = 0 (K-pad for post layer 1)
    act[row * 64 + lane] = (_Float16)((lane < 8) ? zv : 0.f);
  }

  // ---- post layer 1: K=8(pad32), N=32 -> relu ----
  {
    v8f t[2];
#pragma unroll
    for (int nt = 0; nt < 2; ++nt) {
      v8f acc = {};
      acc = wmma32(load_frag(act, 64, 0, lane),
                   load_frag(s_p1t + nt * 16 * 32, 32, 0, lane), acc);
      t[nt] = acc;
    }
#pragma unroll
    for (int nt = 0; nt < 2; ++nt)
#pragma unroll
      for (int v = 0; v < 8; ++v) {
        float val = t[nt][v] + s_bp1[nt * 16 + nl];
        act[(mb + v) * 64 + nt * 16 + nl] = (_Float16)fmaxf(val, 0.f);
      }
  }

  // ---- post layer 2: K=32, N=16 -> relu (zero K-pad cols 16..31) ----
  {
    v8f acc = {};
    acc = wmma32(load_frag(act, 64, 0, lane), load_frag(s_p2t, 32, 0, lane), acc);
#pragma unroll
    for (int v = 0; v < 8; ++v) {
      float val = acc[v] + s_bp2[nl];
      act[(mb + v) * 64 + nl]      = (_Float16)fmaxf(val, 0.f);
      act[(mb + v) * 64 + 16 + nl] = (_Float16)0.f;
    }
  }

  // ---- post layer 3: K=16(pad32), N=5(pad16) -> global out ----
  {
    v8f acc = {};
    acc = wmma32(load_frag(act, 64, 0, lane), load_frag(s_p3t, 32, 0, lane), acc);
#pragma unroll
    for (int v = 0; v < 8; ++v) {
      if (nl < 5) out[(row0 + mb + v) * 5 + nl] = acc[v] + s_bp3[nl];
    }
  }
}

extern "C" void kernel_launch(void* const* d_in, const int* in_sizes, int n_in,
                              void* d_out, int out_size, void* d_ws, size_t ws_size,
                              hipStream_t stream) {
  const float* x   = (const float*)d_in[0];
  const float* w1  = (const float*)d_in[1];
  const float* b1  = (const float*)d_in[2];
  const float* w2  = (const float*)d_in[3];
  const float* b2  = (const float*)d_in[4];
  const float* w3  = (const float*)d_in[5];
  const float* b3  = (const float*)d_in[6];
  const float* qw  = (const float*)d_in[7];
  const float* p1  = (const float*)d_in[8];
  const float* bp1 = (const float*)d_in[9];
  const float* p2  = (const float*)d_in[10];
  const float* bp2 = (const float*)d_in[11];
  const float* p3  = (const float*)d_in[12];
  const float* bp3 = (const float*)d_in[13];

  const int n_rows = in_sizes[0] / 41;   // 16384
  const int blocks = n_rows / 64;        // 64 rows per block (4 waves x 16 rows)

  hqc_fused_kernel<<<blocks, 128, 0, stream>>>(x, w1, b1, w2, b2, w3, b3, qw,
                                               p1, bp1, p2, bp2, p3, bp3,
                                               (float*)d_out);
}
